// MultiHeadAttention_60129542563
// MI455X (gfx1250) — matrix operand order repaired and checked by reading, not compiled, not run on hardware
//
#include <hip/hip_runtime.h>
#include <hip/hip_bf16.h>

// ---------------------------------------------------------------------------
// Causal MHA forward for MI455X (gfx1250, wave32, WMMA).
//   out = softmax(mask(Q K^T / sqrt(D))) V  @ Wo ; also returns K, V heads.
// bf16 operands, f32 accumulation via v_wmma_f32_16x16x32_bf16.
// Flash-attention style: scores never touch HBM.
// Global->LDS staging uses GLOBAL_LOAD_ASYNC_TO_LDS_B128 (ASYNCcnt) with LDS
// double-buffering when the builtin exists; otherwise a register-pipelined
// synchronous path that still overlaps HBM latency with WMMA compute.
// Workspace layout (bytes), total ~92 MB:
//   xbf   [8192,1024] bf16   @ 0
//   wqkvT [3072,1024] bf16   @ 16 MiB
//   woT   [1024,1024] bf16   @ 22 MiB
//   qbf   [B,H,T,D]   bf16   @ 24 MiB   (pre-scaled by 1/sqrt(D))
//   kbf   [B,H,T,D]   bf16   @ 40 MiB
//   vbf   [B,H,T,D]   bf16   @ 56 MiB
//   ctxbf [8192,1024] bf16   @ 72 MiB
// ---------------------------------------------------------------------------

typedef __bf16 bf16_t;
typedef __bf16 v8bf  __attribute__((ext_vector_type(8)));
typedef __bf16 v16bf __attribute__((ext_vector_type(16)));
typedef float  v8f   __attribute__((ext_vector_type(8)));
typedef int    v4i_t __attribute__((ext_vector_type(4)));

static constexpr int BB = 4, TT = 2048, HH = 16, DD = 64, CC = 1024;
static constexpr int MTOT = BB * TT;      // 8192
static constexpr int NQKV = 3 * CC;       // 3072

#define WMMA_BF16(a, b, c) \
  __builtin_amdgcn_wmma_f32_16x16x32_bf16(false, (a), false, (b), (short)0, (c), false, false)

// ---- async global->LDS (gfx1250) with graceful fallback --------------------
#if __has_builtin(__builtin_amdgcn_global_load_async_to_lds_b128)
#define HAVE_ASYNC_LDS 1
typedef __attribute__((address_space(3))) v4i_t* las_ptr_t;
static __device__ __forceinline__ void cp_async16(void* lds, const void* g) {
  // (global VA, LDS addr, imm offset, imm cpol)
  __builtin_amdgcn_global_load_async_to_lds_b128(
      (v4i_t*)g, (las_ptr_t)(v4i_t*)lds, 0, 0);
}
#if __has_builtin(__builtin_amdgcn_s_wait_asynccnt)
#define WAIT_ASYNC(n) __builtin_amdgcn_s_wait_asynccnt(n)
#else
#define WAIT_ASYNC(n) asm volatile("s_wait_asynccnt %0" ::"n"(n) : "memory")
#endif
#endif

static __device__ __forceinline__ v16bf cat8(v8bf lo, v8bf hi) {
  v16bf r;
#pragma unroll
  for (int i = 0; i < 8; ++i) { r[i] = lo[i]; r[8 + i] = hi[i]; }
  return r;
}
static __device__ __forceinline__ v8f vzero() {
  v8f z;
#pragma unroll
  for (int i = 0; i < 8; ++i) z[i] = 0.0f;
  return z;
}

// --------------------------- elementwise converts ---------------------------

__global__ __launch_bounds__(256) void cvt_f32_bf16_k(const float* __restrict__ s,
                                                      bf16_t* __restrict__ d, int n) {
  int i = (blockIdx.x * 256 + threadIdx.x) * 4;
  if (i + 3 < n) {
    float4 f = *(const float4*)(s + i);
    d[i + 0] = (bf16_t)f.x; d[i + 1] = (bf16_t)f.y;
    d[i + 2] = (bf16_t)f.z; d[i + 3] = (bf16_t)f.w;
  }
}

// wt[n*K + k] = bf16(w[k*N + n])  (weights pre-transposed so WMMA B-fragment
// reads are contiguous along K)
__global__ __launch_bounds__(256) void transpose_f32_bf16_k(const float* __restrict__ w,
                                                            bf16_t* __restrict__ wt,
                                                            int K, int N) {
  long idx = (long)blockIdx.x * 256 + threadIdx.x;
  if (idx < (long)N * K) {
    int n = (int)(idx / K), k = (int)(idx % K);
    wt[idx] = (bf16_t)w[(long)k * N + n];
  }
}

// --------------------------- 128x128 GEMM core ------------------------------
// C[128,128] += A[128,K] * Bt[128,K]^T  (both row-major along K)
// 8 waves: wave m-group = wave>>1 (32 rows), n-group = wave&1 (64 cols).
static __device__ __forceinline__ void gemm_tile_128x128(const bf16_t* __restrict__ A,
                                                         const bf16_t* __restrict__ Bt,
                                                         int K, int m0, int n0,
                                                         v8f acc[2][4]) {
  __shared__ __align__(16) bf16_t As[2][128][32];
  __shared__ __align__(16) bf16_t Bs[2][128][32];

  const int tid  = threadIdx.x;
  const int lane = tid & 31;
  const int wave = tid >> 5;
  const int wm   = wave >> 1;
  const int wn   = wave & 1;

#pragma unroll
  for (int i = 0; i < 2; ++i)
#pragma unroll
    for (int j = 0; j < 4; ++j) acc[i][j] = vzero();

  const int lr = tid >> 1;          // tile row this thread stages (0..127)
  const int lc = (tid & 1) * 16;    // 16-element chunk base within 32-wide slab
  const int l15 = lane & 15;
  const int klo = (lane >> 4) * 8;  // A-fragment K base (ISA 16-bit A layout)
  const int kb  = (lane >> 4) * 16; // unused; B fragment reads use klo like A
  const bf16_t* gA = A  + (long)(m0 + lr) * K + lc;
  const bf16_t* gB = Bt + (long)(n0 + lr) * K + lc;
  const int nk = K >> 5;

  auto compute = [&](int cur) {
    v16bf af[2], bfv[4];
#pragma unroll
    for (int mt = 0; mt < 2; ++mt) {
      int m = wm * 32 + mt * 16 + l15;
      af[mt] = cat8(*(const v8bf*)&As[cur][m][klo], *(const v8bf*)&As[cur][m][klo + 16]);
    }
#pragma unroll
    for (int nt = 0; nt < 4; ++nt) {
      int n = wn * 64 + nt * 16 + l15;
      bfv[nt] = cat8(*(const v8bf*)&Bs[cur][n][klo], *(const v8bf*)&Bs[cur][n][klo + 16]);
    }
#pragma unroll
    for (int mt = 0; mt < 2; ++mt)
#pragma unroll
      for (int nt = 0; nt < 4; ++nt)
        acc[mt][nt] = WMMA_BF16(af[mt], bfv[nt], acc[mt][nt]);
  };

#if defined(HAVE_ASYNC_LDS)
  // Double-buffered async pipeline: 4 b128 async loads per thread per tile.
  cp_async16(&As[0][lr][lc],     gA);
  cp_async16(&As[0][lr][lc + 8], gA + 8);
  cp_async16(&Bs[0][lr][lc],     gB);
  cp_async16(&Bs[0][lr][lc + 8], gB + 8);
  for (int i = 0; i < nk; ++i) {
    const int cur = i & 1;
    if (i + 1 < nk) {
      const bf16_t* nA = gA + (i + 1) * 32;
      const bf16_t* nB = gB + (i + 1) * 32;
      cp_async16(&As[cur ^ 1][lr][lc],     nA);
      cp_async16(&As[cur ^ 1][lr][lc + 8], nA + 8);
      cp_async16(&Bs[cur ^ 1][lr][lc],     nB);
      cp_async16(&Bs[cur ^ 1][lr][lc + 8], nB + 8);
      WAIT_ASYNC(4);   // tile i landed; tile i+1 still in flight
    } else {
      WAIT_ASYNC(0);
    }
    __syncthreads();
    compute(cur);
    __syncthreads();   // readers done before buf[cur] is overwritten (i+2)
  }
#else
  // Register-pipelined synchronous fallback: next tile loads overlap compute.
  uint4 ra0 = *(const uint4*)gA, ra1 = *(const uint4*)(gA + 8);
  uint4 rb0 = *(const uint4*)gB, rb1 = *(const uint4*)(gB + 8);
  for (int i = 0; i < nk; ++i) {
    *(uint4*)&As[0][lr][lc]     = ra0;
    *(uint4*)&As[0][lr][lc + 8] = ra1;
    *(uint4*)&Bs[0][lr][lc]     = rb0;
    *(uint4*)&Bs[0][lr][lc + 8] = rb1;
    if (i + 1 < nk) {
      const bf16_t* nA = gA + (i + 1) * 32;
      const bf16_t* nB = gB + (i + 1) * 32;
      ra0 = *(const uint4*)nA;  ra1 = *(const uint4*)(nA + 8);
      rb0 = *(const uint4*)nB;  rb1 = *(const uint4*)(nB + 8);
    }
    __syncthreads();
    compute(0);
    __syncthreads();
  }
#endif
}

// QKV projection: scatter into head layout. q scaled by 1/sqrt(D)=0.125.
__global__ __launch_bounds__(256) void gemm_qkv_k(const bf16_t* __restrict__ A,
                                                  const bf16_t* __restrict__ Bt,
                                                  float* __restrict__ kout,
                                                  float* __restrict__ vout,
                                                  bf16_t* __restrict__ qbf,
                                                  bf16_t* __restrict__ kbf,
                                                  bf16_t* __restrict__ vbf) {
  const int m0 = blockIdx.y * 128, n0 = blockIdx.x * 128;
  v8f acc[2][4];
  gemm_tile_128x128(A, Bt, CC, m0, n0, acc);

  const int lane = threadIdx.x & 31, wave = threadIdx.x >> 5;
  const int wm = wave >> 1, wn = wave & 1;
  const int half8 = (lane >> 4) * 8, l15 = lane & 15;
#pragma unroll
  for (int mt = 0; mt < 2; ++mt)
#pragma unroll
    for (int nt = 0; nt < 4; ++nt)
#pragma unroll
      for (int r = 0; r < 8; ++r) {
        int m = m0 + wm * 32 + mt * 16 + r + half8;   // row = b*T + t
        int n = n0 + wn * 64 + nt * 16 + l15;         // col in [0,3C)
        float v = acc[mt][nt][r];
        int which = n >> 10;            // 0=q 1=k 2=v
        int hd = n & 1023;
        int bi = m >> 11, t = m & 2047;
        long idx = ((long)(bi * HH + (hd >> 6)) * TT + t) * DD + (hd & 63);
        if (which == 0)      qbf[idx] = (bf16_t)(v * 0.125f);
        else if (which == 1) { kout[idx] = v; kbf[idx] = (bf16_t)v; }
        else                 { vout[idx] = v; vbf[idx] = (bf16_t)v; }
      }
}

// Output projection: out = ctx @ Wo
__global__ __launch_bounds__(256) void gemm_out_k(const bf16_t* __restrict__ A,
                                                  const bf16_t* __restrict__ Bt,
                                                  float* __restrict__ out) {
  const int m0 = blockIdx.y * 128, n0 = blockIdx.x * 128;
  v8f acc[2][4];
  gemm_tile_128x128(A, Bt, CC, m0, n0, acc);

  const int lane = threadIdx.x & 31, wave = threadIdx.x >> 5;
  const int wm = wave >> 1, wn = wave & 1;
  const int half8 = (lane >> 4) * 8, l15 = lane & 15;
#pragma unroll
  for (int mt = 0; mt < 2; ++mt)
#pragma unroll
    for (int nt = 0; nt < 4; ++nt)
#pragma unroll
      for (int r = 0; r < 8; ++r) {
        int m = m0 + wm * 32 + mt * 16 + r + half8;
        int n = n0 + wn * 64 + nt * 16 + l15;
        out[(long)m * CC + n] = acc[mt][nt][r];
      }
}

// --------------------------- flash attention --------------------------------
// grid = (T/128, B*H); block = 256 (8 waves). Wave w owns query rows
// [qt*128 + w*16, +16). Streams 64-key tiles with online softmax.
__global__ __launch_bounds__(256) void attn_k(const bf16_t* __restrict__ qbf,
                                              const bf16_t* __restrict__ kbf,
                                              const bf16_t* __restrict__ vbf,
                                              bf16_t* __restrict__ ctx) {
  __shared__ __align__(16) bf16_t Qs[128][64];    // 16 KB
  __shared__ __align__(16) bf16_t Ks[64][64];     //  8 KB  [key][d]
  __shared__ __align__(16) bf16_t Vts[64][64];    //  8 KB  [d][key] (transposed)
  __shared__ __align__(16) bf16_t Ps[8][16][64];  // 16 KB  wave-private P staging

  const int tid = threadIdx.x, wave = tid >> 5, lane = tid & 31;
  const int l15 = lane & 15, half8 = (lane >> 4) * 8;
  const int klo = half8;                 // A-fragment K base (0 or 8)
  const int qt = blockIdx.x;
  const int bh = blockIdx.y;             // b*H + h
  const long hoff = (long)bh * TT * DD;

  // load Q tile [128,64] (32 bf16 per thread)
  {
    int r = tid >> 1, c = (tid & 1) * 32;
    const bf16_t* src = qbf + hoff + (long)(qt * 128 + r) * DD + c;
#if defined(HAVE_ASYNC_LDS)
    cp_async16(&Qs[r][c],      src);
    cp_async16(&Qs[r][c + 8],  src + 8);
    cp_async16(&Qs[r][c + 16], src + 16);
    cp_async16(&Qs[r][c + 24], src + 24);
#else
    *(uint4*)&Qs[r][c]      = ((const uint4*)src)[0];
    *(uint4*)&Qs[r][c + 8]  = ((const uint4*)src)[1];
    *(uint4*)&Qs[r][c + 16] = ((const uint4*)src)[2];
    *(uint4*)&Qs[r][c + 24] = ((const uint4*)src)[3];
#endif
  }

  v8f cacc[4];
#pragma unroll
  for (int i = 0; i < 4; ++i) cacc[i] = vzero();
  float mrow[8], lrow[8];
#pragma unroll
  for (int r = 0; r < 8; ++r) { mrow[r] = -1e30f; lrow[r] = 0.0f; }

  const int ktEnd = qt * 2 + 1;  // last key tile touching this query block
  for (int kt = 0; kt <= ktEnd; ++kt) {
    __syncthreads();             // prior readers done before restaging Ks/Vts
    {  // cooperative K tile (16 bf16/thread) + transposed V tile
      int r = tid >> 2, c = (tid & 3) * 16;
      const bf16_t* ks = kbf + hoff + (long)(kt * 64 + r) * DD + c;
#if defined(HAVE_ASYNC_LDS)
      cp_async16(&Ks[r][c],     ks);
      cp_async16(&Ks[r][c + 8], ks + 8);
#else
      *(uint4*)&Ks[r][c]     = ((const uint4*)ks)[0];
      *(uint4*)&Ks[r][c + 8] = ((const uint4*)ks)[1];
#endif
      const bf16_t* vs = vbf + hoff + (long)(kt * 64 + r) * DD + c;
      v8bf v0 = ((const v8bf*)vs)[0], v1 = ((const v8bf*)vs)[1];
#pragma unroll
      for (int i = 0; i < 8; ++i) { Vts[c + i][r] = v0[i]; Vts[c + 8 + i][r] = v1[i]; }
#if defined(HAVE_ASYNC_LDS)
      WAIT_ASYNC(0);             // covers Q (first iter) + this K tile
#endif
    }
    __syncthreads();

    // S = Q_wave(16x64) * K^T  -> 4 n-tiles, 2 K-steps of 32
    v8f sac[4];
#pragma unroll
    for (int i = 0; i < 4; ++i) sac[i] = vzero();
#pragma unroll
    for (int dk = 0; dk < 64; dk += 32) {
      int m = wave * 16 + l15;
      v16bf a = cat8(*(const v8bf*)&Qs[m][dk + klo],
                     *(const v8bf*)&Qs[m][dk + klo + 16]);
      int kb = dk + (lane >> 4) * 16;
#pragma unroll
      for (int nt = 0; nt < 4; ++nt) {
        int n = nt * 16 + l15;
        v16bf b = cat8(*(const v8bf*)&Ks[n][dk + klo], *(const v8bf*)&Ks[n][dk + klo + 16]);
        sac[nt] = WMMA_BF16(a, b, sac[nt]);
      }
    }

    // causal mask + online softmax (rows live in one 16-lane half -> shfl_xor)
#pragma unroll
    for (int r = 0; r < 8; ++r) {
      int gq = qt * 128 + wave * 16 + r + half8;
      float rmax = -1e30f;
#pragma unroll
      for (int nt = 0; nt < 4; ++nt) {
        int gk = kt * 64 + nt * 16 + l15;
        float s = sac[nt][r];
        if (gk > gq) s = -1e30f;
        sac[nt][r] = s;
        rmax = fmaxf(rmax, s);
      }
#pragma unroll
      for (int off = 8; off >= 1; off >>= 1)
        rmax = fmaxf(rmax, __shfl_xor(rmax, off, 32));
      float mnew = fmaxf(mrow[r], rmax);
      float fac = __expf(mrow[r] - mnew);
      float psum = 0.0f;
#pragma unroll
      for (int nt = 0; nt < 4; ++nt) {
        float p = __expf(sac[nt][r] - mnew);
        psum += p;
        Ps[wave][r + half8][nt * 16 + l15] = (bf16_t)p;
      }
#pragma unroll
      for (int off = 8; off >= 1; off >>= 1) psum += __shfl_xor(psum, off, 32);
      lrow[r] = lrow[r] * fac + psum;
      mrow[r] = mnew;
#pragma unroll
      for (int dt = 0; dt < 4; ++dt) cacc[dt][r] *= fac;
    }

    // ctx += P(16x64) * V(64x64)  (Ps is wave-private; in-wave LDS is ordered)
#pragma unroll
    for (int kk = 0; kk < 64; kk += 32) {
      v16bf a = cat8(*(const v8bf*)&Ps[wave][l15][kk + klo],
                     *(const v8bf*)&Ps[wave][l15][kk + klo + 16]);
      int kb = kk + (lane >> 4) * 16;
#pragma unroll
      for (int dt = 0; dt < 4; ++dt) {
        int n = dt * 16 + l15;
        v16bf b = cat8(*(const v8bf*)&Vts[n][kk + klo], *(const v8bf*)&Vts[n][kk + klo + 16]);
        cacc[dt] = WMMA_BF16(a, b, cacc[dt]);
      }
    }
  }

  // normalize and store ctx in [B,T,C] layout for the output GEMM
  const int bb = bh >> 4, h = bh & 15;
#pragma unroll
  for (int dt = 0; dt < 4; ++dt)
#pragma unroll
    for (int r = 0; r < 8; ++r) {
      int t = qt * 128 + wave * 16 + r + half8;
      int d = dt * 16 + l15;
      float v = cacc[dt][r] / lrow[r];
      ctx[((long)(bb * TT + t)) * CC + h * 64 + d] = (bf16_t)v;
    }
}

// ------------------------------- launch -------------------------------------

extern "C" void kernel_launch(void* const* d_in, const int* in_sizes, int n_in,
                              void* d_out, int out_size, void* d_ws, size_t ws_size,
                              hipStream_t stream) {
  (void)in_sizes; (void)n_in; (void)out_size; (void)ws_size;

  const float* x     = (const float*)d_in[0];
  const float* w_qkv = (const float*)d_in[1];
  const float* w_o   = (const float*)d_in[2];
  // d_in[3] = causal mask (computed analytically in-kernel)

  float* out  = (float*)d_out;
  float* kout = out + (size_t)MTOT * CC;
  float* vout = kout + (size_t)MTOT * CC;

  char* ws = (char*)d_ws;
  bf16_t* xbf   = (bf16_t*)(ws);
  bf16_t* wqkvT = (bf16_t*)(ws + (size_t)16 * 1024 * 1024);
  bf16_t* woT   = (bf16_t*)(ws + (size_t)22 * 1024 * 1024);
  bf16_t* qbf   = (bf16_t*)(ws + (size_t)24 * 1024 * 1024);
  bf16_t* kbf   = (bf16_t*)(ws + (size_t)40 * 1024 * 1024);
  bf16_t* vbf   = (bf16_t*)(ws + (size_t)56 * 1024 * 1024);
  bf16_t* ctxbf = (bf16_t*)(ws + (size_t)72 * 1024 * 1024);

  cvt_f32_bf16_k<<<(MTOT * CC) / 1024, 256, 0, stream>>>(x, xbf, MTOT * CC);
  transpose_f32_bf16_k<<<(NQKV * CC) / 256, 256, 0, stream>>>(w_qkv, wqkvT, CC, NQKV);
  transpose_f32_bf16_k<<<(CC * CC) / 256, 256, 0, stream>>>(w_o, woT, CC, CC);

  gemm_qkv_k<<<dim3(NQKV / 128, MTOT / 128), 256, 0, stream>>>(
      xbf, wqkvT, kout, vout, qbf, kbf, vbf);

  attn_k<<<dim3(TT / 128, BB * HH), 256, 0, stream>>>(qbf, kbf, vbf, ctxbf);

  gemm_out_k<<<dim3(CC / 128, MTOT / 128), 256, 0, stream>>>(ctxbf, woT, out);
}
